// MambaFeatureFusion_18064632447027
// MI455X (gfx1250) — matrix operand order repaired and checked by reading, not compiled, not run on hardware
//
#include <hip/hip_runtime.h>
#include <hip/hip_bf16.h>
#include <stdint.h>

// ---- problem dims ----
#define B_    4
#define L_    2048
#define DM    258           // d_model
#define DI    516           // d_inner
#define DS_   16            // d_state
#define DCONV 4
#define DTR   17            // dt_rank
#define XDBL_W (DTR + 2 * DS_)   // 49
#define MTOT  (B_ * L_)          // 8192

// chunked scan
#define NCHUNK 64
#define CLEN   (L_ / NCHUNK)     // 32

// ---- WMMA-padded dims (K mult of 32; N mult of 64 for 32x64 wave tiles) ----
#define KP_IN  288   // pad(DM=258)
#define NP_IN  1088  // pad(2*DI=1032)
#define KP_XI  544   // pad(DI=516)
#define NP_XP  64    // pad(49)
#define KP_DT  32    // pad(17)
#define NP_DT  576   // pad(516)
#define NP_OUT 320   // pad(258)

typedef __attribute__((ext_vector_type(16))) __bf16 v16bf;
typedef __attribute__((ext_vector_type(8)))  float  v8f;

union FragAB { v16bf v; uint4 q[2]; };

// -------------------------------------------------------------------------
// GEMM: C(MxN) = A(MxKp) * W(NxKp)^T (+ optional residual), bf16 in, f32 acc.
// 32x64 output per wave: 2 A-frags x 4 B-frags -> 8 WMMAs per K-step
// (0.75 fragment loads per WMMA instead of 2).
// -------------------------------------------------------------------------
__global__ __launch_bounds__(256)
void wmma_gemm_bf16(const __bf16* __restrict__ A, int lda,
                    const __bf16* __restrict__ W, int ldw,
                    float* __restrict__ C, int ldc,
                    int tilesM, int tilesN, int N, int Kp,
                    const float* __restrict__ residual)
{
    const int lane = threadIdx.x & 31;
    const int wave = threadIdx.x >> 5;
    const int tile = blockIdx.x * 8 + wave;
    if (tile >= tilesM * tilesN) return;     // wave-uniform: EXEC stays all-ones

    const int tn = tile % tilesN;
    const int tm = tile / tilesN;
    const int m0 = tm * 32;
    const int n0 = tn * 64;

    const int half = lane >> 4;              // 0 or 1
    const int l16  = lane & 15;

    // A lane row = l16; per K-step slice {half*16 + 0..15} (contiguous 32B)
    const __bf16* Ar0 = A + (size_t)(m0 + l16) * lda + half * 16;
    const __bf16* Ar1 = Ar0 + (size_t)16 * lda;
    // B lane col = l16; per K-step slice {half*16 + 0..15} (contiguous 32B)
    const __bf16* Br  = W + (size_t)(n0 + l16) * ldw + half * 16;
    const size_t  bstep = (size_t)16 * ldw;

    v8f acc[2][4];
    #pragma unroll
    for (int mi = 0; mi < 2; ++mi)
        #pragma unroll
        for (int ni = 0; ni < 4; ++ni)
            acc[mi][ni] = (v8f){0.f, 0.f, 0.f, 0.f, 0.f, 0.f, 0.f, 0.f};

    for (int k0 = 0; k0 < Kp; k0 += 32) {
        FragAB a0, a1, b0, b1, b2, b3;
        a0.q[0] = *reinterpret_cast<const uint4*>(Ar0 + k0);
        a0.q[1] = *reinterpret_cast<const uint4*>(Ar0 + k0 + 8);
        a1.q[0] = *reinterpret_cast<const uint4*>(Ar1 + k0);
        a1.q[1] = *reinterpret_cast<const uint4*>(Ar1 + k0 + 8);
        b0.q[0] = *reinterpret_cast<const uint4*>(Br + 0 * bstep + k0);
        b0.q[1] = *reinterpret_cast<const uint4*>(Br + 0 * bstep + k0 + 8);
        b1.q[0] = *reinterpret_cast<const uint4*>(Br + 1 * bstep + k0);
        b1.q[1] = *reinterpret_cast<const uint4*>(Br + 1 * bstep + k0 + 8);
        b2.q[0] = *reinterpret_cast<const uint4*>(Br + 2 * bstep + k0);
        b2.q[1] = *reinterpret_cast<const uint4*>(Br + 2 * bstep + k0 + 8);
        b3.q[0] = *reinterpret_cast<const uint4*>(Br + 3 * bstep + k0);
        b3.q[1] = *reinterpret_cast<const uint4*>(Br + 3 * bstep + k0 + 8);

        acc[0][0] = __builtin_amdgcn_wmma_f32_16x16x32_bf16(false, a0.v, false, b0.v, (short)0, acc[0][0], false, false);
        acc[0][1] = __builtin_amdgcn_wmma_f32_16x16x32_bf16(false, a0.v, false, b1.v, (short)0, acc[0][1], false, false);
        acc[0][2] = __builtin_amdgcn_wmma_f32_16x16x32_bf16(false, a0.v, false, b2.v, (short)0, acc[0][2], false, false);
        acc[0][3] = __builtin_amdgcn_wmma_f32_16x16x32_bf16(false, a0.v, false, b3.v, (short)0, acc[0][3], false, false);
        acc[1][0] = __builtin_amdgcn_wmma_f32_16x16x32_bf16(false, a1.v, false, b0.v, (short)0, acc[1][0], false, false);
        acc[1][1] = __builtin_amdgcn_wmma_f32_16x16x32_bf16(false, a1.v, false, b1.v, (short)0, acc[1][1], false, false);
        acc[1][2] = __builtin_amdgcn_wmma_f32_16x16x32_bf16(false, a1.v, false, b2.v, (short)0, acc[1][2], false, false);
        acc[1][3] = __builtin_amdgcn_wmma_f32_16x16x32_bf16(false, a1.v, false, b3.v, (short)0, acc[1][3], false, false);
    }

    #pragma unroll
    for (int ni = 0; ni < 4; ++ni) {
        const int n = n0 + ni * 16 + l16;
        if (n < N) {
            #pragma unroll
            for (int mi = 0; mi < 2; ++mi) {
                #pragma unroll
                for (int r = 0; r < 8; ++r) {
                    const int m = m0 + mi * 16 + half * 8 + r;
                    const size_t idx = (size_t)m * ldc + n;
                    float v = acc[mi][ni][r];
                    if (residual) v += residual[idx];
                    C[idx] = v;
                }
            }
        }
    }
}

// rmsnorm over d_model, write bf16 zero-padded (M x KP_IN) operand
__global__ __launch_bounds__(256)
void rmsnorm_bf16_kernel(const float* __restrict__ x, const float* __restrict__ w,
                         __bf16* __restrict__ out)
{
    __shared__ float red[256];
    const int m = blockIdx.x;
    const int tid = threadIdx.x;
    const float* xr = x + (size_t)m * DM;
    float ss = 0.f;
    for (int i = tid; i < DM; i += 256) { float v = xr[i]; ss += v * v; }
    red[tid] = ss;
    __syncthreads();
    for (int s = 128; s > 0; s >>= 1) {
        if (tid < s) red[tid] += red[tid + s];
        __syncthreads();
    }
    const float rs = rsqrtf(red[0] / (float)DM + 1e-5f);
    __bf16* orow = out + (size_t)m * KP_IN;
    for (int i = tid; i < KP_IN; i += 256) {
        float v = (i < DM) ? xr[i] * rs * w[i] : 0.f;
        orow[i] = (__bf16)v;
    }
}

// final: out = fuse + rmsnorm(x, w)
__global__ __launch_bounds__(256)
void final_kernel(const float* __restrict__ x, const float* __restrict__ w,
                  const float* __restrict__ fuse, float* __restrict__ out)
{
    __shared__ float red[256];
    const int m = blockIdx.x;
    const int tid = threadIdx.x;
    const float* xr = x + (size_t)m * DM;
    float ss = 0.f;
    for (int i = tid; i < DM; i += 256) { float v = xr[i]; ss += v * v; }
    red[tid] = ss;
    __syncthreads();
    for (int s = 128; s > 0; s >>= 1) {
        if (tid < s) red[tid] += red[tid + s];
        __syncthreads();
    }
    const float rs = rsqrtf(red[0] / (float)DM + 1e-5f);
    for (int i = tid; i < DM; i += 256)
        out[(size_t)m * DM + i] = fuse[(size_t)m * DM + i] + xr[i] * rs * w[i];
}

// fp32 (Nn x Kk) weights -> bf16 (Np x Kp) zero-padded
__global__ void pack_w_bf16(const float* __restrict__ src, __bf16* __restrict__ dst,
                            int Nn, int Kk, int Kp, int total)
{
    const int t = blockIdx.x * blockDim.x + threadIdx.x;
    if (t >= total) return;
    const int n = t / Kp, k = t % Kp;
    const float v = (n < Nn && k < Kk) ? src[(size_t)n * Kk + k] : 0.f;
    dst[t] = (__bf16)v;
}

// causal depthwise conv (k=4) + bias + SiLU; writes fp32 u and bf16 padded operand
__global__ void conv_silu_kernel(const float* __restrict__ xz,
                                 const float* __restrict__ cw,
                                 const float* __restrict__ cb,
                                 float* __restrict__ xi_f,
                                 __bf16* __restrict__ xi_bf)
{
    const int t = blockIdx.x * blockDim.x + threadIdx.x;
    if (t >= MTOT * KP_XI) return;
    const int m = t / KP_XI, c = t % KP_XI;
    if (c >= DI) { xi_bf[t] = (__bf16)0.f; return; }
    const int b = m / L_, l = m % L_;
    float s = cb[c];
    #pragma unroll
    for (int j = 0; j < DCONV; ++j) {
        const int ll = l - (DCONV - 1) + j;
        if (ll >= 0)
            s += cw[c * DCONV + j] * xz[((size_t)(b * L_ + ll)) * (2 * DI) + c];
    }
    const float v = s / (1.f + __expf(-s));
    xi_f[(size_t)m * DI + c] = v;
    xi_bf[t] = (__bf16)v;
}

// first DTR columns of x_dbl -> bf16 (M x KP_DT) zero-padded
__global__ void stage_dtlo_kernel(const float* __restrict__ xdbl, __bf16* __restrict__ dst)
{
    const int t = blockIdx.x * blockDim.x + threadIdx.x;
    if (t >= MTOT * KP_DT) return;
    const int m = t / KP_DT, r = t % KP_DT;
    dst[t] = (__bf16)((r < DTR) ? xdbl[(size_t)m * XDBL_W + r] : 0.f);
}

// dt = softplus(dt_raw + bias)
__global__ void softplus_kernel(float* __restrict__ dt, const float* __restrict__ dtb)
{
    const int t = blockIdx.x * blockDim.x + threadIdx.x;
    if (t >= MTOT * DI) return;
    const int d = t % DI;
    const float v = dt[t] + dtb[d];
    dt[t] = (v > 20.f) ? v : log1pf(__expf(v));
}

__global__ void zero_ypad_kernel(__bf16* __restrict__ y_bf)
{
    const int t = blockIdx.x * blockDim.x + threadIdx.x;
    const int padw = KP_XI - DI;
    if (t >= MTOT * padw) return;
    const int m = t / padw, c = DI + t % padw;
    y_bf[(size_t)m * KP_XI + c] = (__bf16)0.f;
}

// ---- chunked parallel scan --------------------------------------------------
// Recurrence h <- a*h + v is associative: chunk aggregate is (Prod a, folded v).
// Aggregate layout (d-fastest, coalesced): CA(b,chunk,s,d) =
//   ((b*NCHUNK + chunk)*DS_ + s)*DI + d

// Pass 1: per-(b,d,chunk) aggregates: Ap = prod(a_l), Bg = sum folded(du*B).
__global__ __launch_bounds__(256)
void scan_pass1_kernel(const float* __restrict__ dt, const float* __restrict__ u,
                       const float* __restrict__ xdbl, const float* __restrict__ A_log,
                       float* __restrict__ chunkA, float* __restrict__ chunkB)
{
    const int t = blockIdx.x * blockDim.x + threadIdx.x;
    if (t >= B_ * NCHUNK * DI) return;
    const int d = t % DI;
    const int r = t / DI;
    const int chunk = r % NCHUNK;
    const int b = r / NCHUNK;

    float Aneg[DS_], Ap[DS_], Bg[DS_];
    #pragma unroll
    for (int s = 0; s < DS_; ++s) {
        Aneg[s] = -__expf(A_log[d * DS_ + s]);
        Ap[s] = 1.f;
        Bg[s] = 0.f;
    }

    const int l0 = chunk * CLEN;
    for (int l = l0; l < l0 + CLEN; ++l) {
        const size_t m = (size_t)b * L_ + l;
        const float dtv = dt[m * DI + d];
        const float du  = dtv * u[m * DI + d];
        const float* bc = xdbl + m * XDBL_W;
        #pragma unroll
        for (int s = 0; s < DS_; ++s) {
            const float a = __expf(dtv * Aneg[s]);
            Bg[s] = a * Bg[s] + du * bc[DTR + s];
            Ap[s] *= a;
        }
    }
    const size_t base = ((size_t)(b * NCHUNK + chunk) * DS_) * DI + d;
    #pragma unroll
    for (int s = 0; s < DS_; ++s) {
        chunkA[base + (size_t)s * DI] = Ap[s];
        chunkB[base + (size_t)s * DI] = Bg[s];
    }
}

// Pass 2: serial compose over 64 chunk boundaries per (b,d).
// Rewrites chunkB in place with the INCOMING state of each chunk.
__global__ __launch_bounds__(256)
void scan_pass2_kernel(float* __restrict__ chunkA, float* __restrict__ chunkB)
{
    const int t = blockIdx.x * blockDim.x + threadIdx.x;
    if (t >= B_ * DI) return;
    const int b = t / DI, d = t % DI;

    float h[DS_];
    #pragma unroll
    for (int s = 0; s < DS_; ++s) h[s] = 0.f;

    for (int chunk = 0; chunk < NCHUNK; ++chunk) {
        const size_t base = ((size_t)(b * NCHUNK + chunk) * DS_) * DI + d;
        #pragma unroll
        for (int s = 0; s < DS_; ++s) {
            const size_t idx = base + (size_t)s * DI;
            const float a  = chunkA[idx];
            const float bv = chunkB[idx];
            chunkB[idx] = h[s];          // incoming state for this chunk
            h[s] = a * h[s] + bv;
        }
    }
}

// Pass 3: replay each chunk from its incoming state; fuse D-skip + SiLU(z) gate.
__global__ __launch_bounds__(256)
void scan_pass3_kernel(const float* __restrict__ dt, const float* __restrict__ u,
                       const float* __restrict__ xdbl, const float* __restrict__ xz,
                       const float* __restrict__ A_log, const float* __restrict__ Dp,
                       const float* __restrict__ chunkB, __bf16* __restrict__ y_bf)
{
    const int t = blockIdx.x * blockDim.x + threadIdx.x;
    if (t >= B_ * NCHUNK * DI) return;
    const int d = t % DI;
    const int r = t / DI;
    const int chunk = r % NCHUNK;
    const int b = r / NCHUNK;

    float Aneg[DS_], h[DS_];
    const size_t base = ((size_t)(b * NCHUNK + chunk) * DS_) * DI + d;
    #pragma unroll
    for (int s = 0; s < DS_; ++s) {
        Aneg[s] = -__expf(A_log[d * DS_ + s]);
        h[s] = chunkB[base + (size_t)s * DI];
    }
    const float Dd = Dp[d];

    const int l0 = chunk * CLEN;
    for (int l = l0; l < l0 + CLEN; ++l) {
        const size_t m = (size_t)b * L_ + l;
        const float dtv = dt[m * DI + d];
        const float uv  = u[m * DI + d];
        const float du  = dtv * uv;
        const float* bc = xdbl + m * XDBL_W;
        float y = 0.f;
        #pragma unroll
        for (int s = 0; s < DS_; ++s) {
            const float a = __expf(dtv * Aneg[s]);
            h[s] = a * h[s] + du * bc[DTR + s];
            y += h[s] * bc[DTR + DS_ + s];
        }
        const float z = xz[m * (2 * DI) + DI + d];
        const float g = z / (1.f + __expf(-z));
        y_bf[m * KP_XI + d] = (__bf16)((y + uv * Dd) * g);
    }
}

extern "C" void kernel_launch(void* const* d_in, const int* in_sizes, int n_in,
                              void* d_out, int out_size, void* d_ws, size_t ws_size,
                              hipStream_t stream)
{
    const float* fuse   = (const float*)d_in[0];
    const float* norm_w = (const float*)d_in[1];
    const float* in_w   = (const float*)d_in[2];
    const float* conv_w = (const float*)d_in[3];
    const float* conv_b = (const float*)d_in[4];
    const float* xp_w   = (const float*)d_in[5];
    const float* dt_w   = (const float*)d_in[6];
    const float* dt_b   = (const float*)d_in[7];
    const float* A_log  = (const float*)d_in[8];
    const float* Dp     = (const float*)d_in[9];
    const float* out_w  = (const float*)d_in[10];
    const float* fnw    = (const float*)d_in[11];

    char* base = (char*)d_ws;
    size_t off = 0;
    auto alloc = [&](size_t bytes) -> void* {
        void* p = base + off;
        off = (off + bytes + 255) & ~(size_t)255;
        return p;
    };
    float*  x_cur   = (float*) alloc((size_t)MTOT * DM * 4);
    __bf16* xn_bf   = (__bf16*)alloc((size_t)MTOT * KP_IN * 2);
    float*  xz      = (float*) alloc((size_t)MTOT * 2 * DI * 4);
    float*  xi_f    = (float*) alloc((size_t)MTOT * DI * 4);
    __bf16* xi_bf   = (__bf16*)alloc((size_t)MTOT * KP_XI * 2);
    float*  xdbl    = (float*) alloc((size_t)MTOT * XDBL_W * 4);
    __bf16* dtlo_bf = (__bf16*)alloc((size_t)MTOT * KP_DT * 2);
    float*  dt_f    = (float*) alloc((size_t)MTOT * DI * 4);
    __bf16* y_bf    = (__bf16*)alloc((size_t)MTOT * KP_XI * 2);
    float*  chunkA  = (float*) alloc((size_t)B_ * NCHUNK * DS_ * DI * 4);
    float*  chunkB  = (float*) alloc((size_t)B_ * NCHUNK * DS_ * DI * 4);
    __bf16* w_in    = (__bf16*)alloc((size_t)NP_IN  * KP_IN * 2);
    __bf16* w_xp    = (__bf16*)alloc((size_t)NP_XP  * KP_XI * 2);
    __bf16* w_dt    = (__bf16*)alloc((size_t)NP_DT  * KP_DT * 2);
    __bf16* w_out   = (__bf16*)alloc((size_t)NP_OUT * KP_XI * 2);

    hipMemcpyAsync(x_cur, fuse, (size_t)MTOT * DM * 4, hipMemcpyDeviceToDevice, stream);

    for (int i = 0; i < 4; ++i) {
        { int tot = NP_IN * KP_IN;
          pack_w_bf16<<<(tot + 255) / 256, 256, 0, stream>>>(
              in_w + (size_t)i * 2 * DI * DM, w_in, 2 * DI, DM, KP_IN, tot); }
        { int tot = NP_XP * KP_XI;
          pack_w_bf16<<<(tot + 255) / 256, 256, 0, stream>>>(
              xp_w + (size_t)i * XDBL_W * DI, w_xp, XDBL_W, DI, KP_XI, tot); }
        { int tot = NP_DT * KP_DT;
          pack_w_bf16<<<(tot + 255) / 256, 256, 0, stream>>>(
              dt_w + (size_t)i * DI * DTR, w_dt, DI, DTR, KP_DT, tot); }
        { int tot = NP_OUT * KP_XI;
          pack_w_bf16<<<(tot + 255) / 256, 256, 0, stream>>>(
              out_w + (size_t)i * DM * DI, w_out, DM, DI, KP_XI, tot); }

        rmsnorm_bf16_kernel<<<MTOT, 256, 0, stream>>>(x_cur, norm_w + (size_t)i * DM, xn_bf);

        { // in_proj: (M x 258) * (1032 x 258)^T -> xz (M x 1032)
          int tilesM = MTOT / 32, tilesN = NP_IN / 64;
          wmma_gemm_bf16<<<(tilesM * tilesN + 7) / 8, 256, 0, stream>>>(
              xn_bf, KP_IN, w_in, KP_IN, xz, 2 * DI, tilesM, tilesN, 2 * DI, KP_IN, nullptr); }

        { int tot = MTOT * KP_XI;
          conv_silu_kernel<<<(tot + 255) / 256, 256, 0, stream>>>(
              xz, conv_w + (size_t)i * DI * DCONV, conv_b + (size_t)i * DI, xi_f, xi_bf); }

        { // x_proj: (M x 516) * (49 x 516)^T -> xdbl (M x 49)
          int tilesM = MTOT / 32, tilesN = NP_XP / 64;
          wmma_gemm_bf16<<<(tilesM * tilesN + 7) / 8, 256, 0, stream>>>(
              xi_bf, KP_XI, w_xp, KP_XI, xdbl, XDBL_W, tilesM, tilesN, XDBL_W, KP_XI, nullptr); }

        { int tot = MTOT * KP_DT;
          stage_dtlo_kernel<<<(tot + 255) / 256, 256, 0, stream>>>(xdbl, dtlo_bf); }

        { // dt_proj: (M x 17) * (516 x 17)^T -> dt_f (M x 516)
          int tilesM = MTOT / 32, tilesN = NP_DT / 64;
          wmma_gemm_bf16<<<(tilesM * tilesN + 7) / 8, 256, 0, stream>>>(
              dtlo_bf, KP_DT, w_dt, KP_DT, dt_f, DI, tilesM, tilesN, DI, KP_DT, nullptr); }

        { int tot = MTOT * DI;
          softplus_kernel<<<(tot + 255) / 256, 256, 0, stream>>>(dt_f, dt_b + (size_t)i * DI); }

        { int tot = MTOT * (KP_XI - DI);
          zero_ypad_kernel<<<(tot + 255) / 256, 256, 0, stream>>>(y_bf); }

        { int tot = B_ * NCHUNK * DI;      // 132K threads
          scan_pass1_kernel<<<(tot + 255) / 256, 256, 0, stream>>>(
              dt_f, xi_f, xdbl, A_log + (size_t)i * DI * DS_, chunkA, chunkB); }
        scan_pass2_kernel<<<(B_ * DI + 255) / 256, 256, 0, stream>>>(chunkA, chunkB);
        { int tot = B_ * NCHUNK * DI;
          scan_pass3_kernel<<<(tot + 255) / 256, 256, 0, stream>>>(
              dt_f, xi_f, xdbl, xz, A_log + (size_t)i * DI * DS_, Dp + (size_t)i * DI,
              chunkB, y_bf); }

        { // out_proj + residual: x_cur = x_cur + (M x 516) * (258 x 516)^T
          int tilesM = MTOT / 32, tilesN = NP_OUT / 64;
          wmma_gemm_bf16<<<(tilesM * tilesN + 7) / 8, 256, 0, stream>>>(
              y_bf, KP_XI, w_out, KP_XI, x_cur, DM, tilesM, tilesN, DM, KP_XI, x_cur); }
    }

    final_kernel<<<MTOT, 256, 0, stream>>>(x_cur, fnw, fuse, (float*)d_out);
}
